// FlowBranch_22290880266613
// MI455X (gfx1250) — matrix operand order repaired and checked by reading, not compiled, not run on hardware
//
#include <hip/hip_runtime.h>
#include <cstdint>
#include <cstddef>

typedef __attribute__((ext_vector_type(16))) __bf16 v16bf;
typedef __attribute__((ext_vector_type(8)))  __bf16 v8bf;
typedef __attribute__((ext_vector_type(8)))  float  v8f;

// ============================================================================
// Elementwise helper kernels
// ============================================================================
__global__ void fill_bf16_zero(__bf16* p, long n) {
  long i = (long)blockIdx.x * blockDim.x + threadIdx.x;
  if (i < n) p[i] = (__bf16)0.0f;
}

// NCHW fp32 -> NHWC bf16 (channel-strided buffer, channel offset)
__global__ void nchw_to_nhwc_bf16(const float* __restrict__ in, __bf16* __restrict__ out,
                                  int B, int C, int H, int W, int CS, int chOff) {
  long i = (long)blockIdx.x * blockDim.x + threadIdx.x;
  long n = (long)B * C * H * W;
  if (i >= n) return;
  int x = (int)(i % W);
  int y = (int)((i / W) % H);
  int c = (int)((i / ((long)W * H)) % C);
  int b = (int)(i / ((long)W * H * C));
  out[((long)((b * H + y) * W + x)) * CS + chOff + c] = (__bf16)in[i];
}

// Cost volume: out[p][o] = mean_c f1[b,c,y,x] * f2[b,c,y+dy,x+dx], d=4 (zero pad)
__global__ void corr_kernel(const float* __restrict__ f1, const float* __restrict__ f2,
                            __bf16* __restrict__ out, int B, int C, int H, int W, int CS) {
  long i = (long)blockIdx.x * blockDim.x + threadIdx.x;
  long n = (long)B * H * W * 81;
  if (i >= n) return;
  int o = (int)(i % 81);
  long p = i / 81;
  int x = (int)(p % W);
  int y = (int)((p / W) % H);
  int b = (int)(p / ((long)W * H));
  int dy = o / 9 - 4, dx = o % 9 - 4;
  int yy = y + dy, xx = x + dx;
  float acc = 0.0f;
  if (yy >= 0 && yy < H && xx >= 0 && xx < W) {
    const float* a  = f1 + (long)b * C * H * W + (long)y * W + x;
    const float* bp = f2 + (long)b * C * H * W + (long)yy * W + xx;
    long st = (long)H * W;
    for (int c = 0; c < C; ++c) acc += a[(long)c * st] * bp[(long)c * st];
  }
  out[p * CS + o] = (__bf16)(acc / (float)C);
}

// Bilinear warp (grid_sample, zeros padding, align_corners pixel coords)
__global__ void warp_kernel(const float* __restrict__ x, const float* __restrict__ flow,
                            float* __restrict__ out, int B, int C, int H, int W) {
  long i = (long)blockIdx.x * blockDim.x + threadIdx.x;
  long n = (long)B * C * H * W;
  if (i >= n) return;
  int xw = (int)(i % W);
  int yh = (int)((i / W) % H);
  int c  = (int)((i / ((long)W * H)) % C);
  int b  = (int)(i / ((long)W * H * C));
  float gx = flow[(((long)b * 2 + 0) * H + yh) * W + xw] + (float)xw;
  float gy = flow[(((long)b * 2 + 1) * H + yh) * W + xw] + (float)yh;
  float x0f = floorf(gx), y0f = floorf(gy);
  float wx = gx - x0f, wy = gy - y0f;
  int x0 = (int)x0f, y0 = (int)y0f;
  const float* p = x + ((long)b * C + c) * H * W;
  float v00 = 0.f, v01 = 0.f, v10 = 0.f, v11 = 0.f;
  if (y0 >= 0 && y0 < H) {
    if (x0 >= 0 && x0 < W)         v00 = p[(long)y0 * W + x0];
    if (x0 + 1 >= 0 && x0 + 1 < W) v01 = p[(long)y0 * W + x0 + 1];
  }
  if (y0 + 1 >= 0 && y0 + 1 < H) {
    if (x0 >= 0 && x0 < W)         v10 = p[(long)(y0 + 1) * W + x0];
    if (x0 + 1 >= 0 && x0 + 1 < W) v11 = p[(long)(y0 + 1) * W + x0 + 1];
  }
  out[i] = v00 * (1.f - wx) * (1.f - wy) + v01 * wx * (1.f - wy)
         + v10 * (1.f - wx) * wy + v11 * wx * wy;
}

// align_corners bilinear resize (NCHW fp32), multiplies by `mult`
__global__ void resize_ac_kernel(const float* __restrict__ in, float* __restrict__ out,
                                 int BC, int Hi, int Wi, int Ho, int Wo, float mult) {
  long i = (long)blockIdx.x * blockDim.x + threadIdx.x;
  long n = (long)BC * Ho * Wo;
  if (i >= n) return;
  int x  = (int)(i % Wo);
  int y  = (int)((i / Wo) % Ho);
  int bc = (int)(i / ((long)Wo * Ho));
  int y0 = y; float wy = 0.f;
  if (Hi != Ho) {
    float sc = (float)(Hi - 1) / (float)(Ho - 1);
    float pos = (float)y * sc;
    y0 = (int)floorf(pos);
    if (y0 > Hi - 2) y0 = Hi - 2;
    if (y0 < 0) y0 = 0;
    wy = pos - (float)y0;
  }
  int x0 = x; float wx = 0.f;
  if (Wi != Wo) {
    float sc = (float)(Wi - 1) / (float)(Wo - 1);
    float pos = (float)x * sc;
    x0 = (int)floorf(pos);
    if (x0 > Wi - 2) x0 = Wi - 2;
    if (x0 < 0) x0 = 0;
    wx = pos - (float)x0;
  }
  int y1 = (y0 + 1 < Hi) ? y0 + 1 : Hi - 1;
  int x1 = (x0 + 1 < Wi) ? x0 + 1 : Wi - 1;
  const float* p = in + (long)bc * Hi * Wi;
  float v = (1.f - wy) * ((1.f - wx) * p[(long)y0 * Wi + x0] + wx * p[(long)y0 * Wi + x1])
          +        wy  * ((1.f - wx) * p[(long)y1 * Wi + x0] + wx * p[(long)y1 * Wi + x1]);
  out[i] = v * mult;
}

// ============================================================================
// Weight packing: fp32 (cout,cin,3,3) -> bf16 B-fragments.
// Fragment layout: element t = ((kc*NT + nt)*32 + lane)*16 + h
//   K = kc*32 + (lane>=16 ? 8 : 0) + (h&7) + 2*(h&8)  (K = r*CinP + c, r = dy*3+dx outer)
//   N = nt*16 + (lane & 15)
// ============================================================================
__global__ void pack_weights(const float* __restrict__ w, __bf16* __restrict__ wpk,
                             int Cin, int CinP, int NT, long total) {
  long t = (long)blockIdx.x * blockDim.x + threadIdx.x;
  if (t >= total) return;
  int h    = (int)(t & 15);
  int lane = (int)((t >> 4) & 31);
  long rest = t >> 9;              // kc*NT + nt
  int nt = (int)(rest % NT);
  long kc = rest / NT;
  long K = kc * 32 + ((lane >= 16) ? 8 : 0) + (h & 7) + 2 * (h & 8);
  int n = nt * 16 + (lane & 15);
  int r = (int)(K / CinP);
  int c = (int)(K % CinP);
  float val = (c < Cin) ? w[((long)n * Cin + c) * 9 + r] : 0.0f;
  wpk[t] = (__bf16)val;
}

// ============================================================================
// Implicit-GEMM 3x3 conv, bf16 WMMA, fp32 accumulate.
// 4 waves/block, one 16-pixel M-tile per wave; NT N-tiles fully unrolled.
// A-fragment element (lane,h): M = lane&15, K = (h&7) + 2*(h&8) + 8*(lane>=16)
//   -> two contiguous 8x bf16 (16B) loads per lane per 32-K chunk.
// Zero-padding is branch-free: addresses clamped in-bounds, fragment masked.
// ============================================================================
union U16B { uint4 u; v8bf h; };

template <int NT>
__global__ __launch_bounds__(128)
void conv3x3_wmma(const __bf16* __restrict__ in, int inCS, int inOff,
                  const __bf16* __restrict__ wpk, const float* __restrict__ bias,
                  __bf16* __restrict__ out, int outCS, int outOff,
                  int Bn, int H, int W, int CinP, int leaky) {
  const int lane = threadIdx.x & 31;
  const int mt = blockIdx.x * 4 + (threadIdx.x >> 5);
  const int pixels = Bn * H * W;
  const int mA = mt * 16 + (lane & 15);
  const bool mvalid = (mA < pixels);
  const int mAc = mvalid ? mA : pixels - 1;    // clamp for safe addressing
  int b = mAc / (H * W);
  int rem = mAc - b * (H * W);
  int y = rem / W;
  int x = rem - y * W;

  v8f acc[NT];
#pragma unroll
  for (int i = 0; i < NT; ++i) acc[i] = (v8f){0.f, 0.f, 0.f, 0.f, 0.f, 0.f, 0.f, 0.f};

  const int cpc = CinP >> 5;                   // 32-channel chunks per tap
  const int sub = (lane >= 16) ? 8 : 0;
  const v16bf* wbase = (const v16bf*)wpk;

  for (int r = 0; r < 9; ++r) {
    const int dy = r / 3 - 1, dx = r % 3 - 1;
    const int yy = y + dy, xx = x + dx;
    const bool v = mvalid && (yy >= 0) && (yy < H) && (xx >= 0) && (xx < W);
    const int yyc = (yy < 0) ? 0 : ((yy >= H) ? H - 1 : yy);
    const int xxc = (xx < 0) ? 0 : ((xx >= W) ? W - 1 : xx);
    const unsigned msk = v ? 0xffffffffu : 0u;
    const __bf16* base = in + (long)((b * H + yyc) * W + xxc) * inCS + inOff + sub;
    for (int cb = 0; cb < cpc; ++cb) {
      const uint4* p = (const uint4*)(base + cb * 32);
      U16B lo, hi;
      lo.u = p[0];
      hi.u = p[2];                             // +16 bf16 elements
      lo.u.x &= msk; lo.u.y &= msk; lo.u.z &= msk; lo.u.w &= msk;
      hi.u.x &= msk; hi.u.y &= msk; hi.u.z &= msk; hi.u.w &= msk;
      v16bf a;
#pragma unroll
      for (int i = 0; i < 8; ++i) { a[i] = lo.h[i]; a[i + 8] = hi.h[i]; }
      const long kc = (long)r * cpc + cb;
#pragma unroll
      for (int nt = 0; nt < NT; ++nt) {
        v16bf bf = wbase[(kc * NT + nt) * 32 + lane];
        acc[nt] = __builtin_amdgcn_wmma_f32_16x16x32_bf16(
            false, a, false, bf, (short)0, acc[nt], false, false);
      }
    }
  }

  // D layout: lane 0-15 -> N=lane, M=vg; lane 16-31 -> N=lane-16, M=8+vg
  const int mrow = mt * 16 + ((lane >= 16) ? 8 : 0);
  const int ncol = lane & 15;
#pragma unroll
  for (int nt = 0; nt < NT; ++nt) {
    const int n = nt * 16 + ncol;
    const float bv = bias[n];
#pragma unroll
    for (int vg = 0; vg < 8; ++vg) {
      const int m = mrow + vg;
      if (m < pixels) {
        float val = acc[nt][vg] + bv;
        if (leaky) val = (val > 0.f) ? val : 0.1f * val;
        out[(long)m * outCS + outOff + n] = (__bf16)val;
      }
    }
  }
}

// Flow head: 3x3 conv 96->2, fp32 out, optional residual add (flow = conv + up)
__global__ void pred_conv(const __bf16* __restrict__ D, const float* __restrict__ w,
                          const float* __restrict__ bias, const float* __restrict__ up,
                          float* __restrict__ out, int Bn, int H, int W,
                          int inCS, int inOff, int Cin) {
  long i = (long)blockIdx.x * blockDim.x + threadIdx.x;
  long n = (long)Bn * 2 * H * W;
  if (i >= n) return;
  int x  = (int)(i % W);
  int y  = (int)((i / W) % H);
  int ch = (int)((i / ((long)W * H)) % 2);
  int b  = (int)(i / ((long)W * H * 2));
  float acc = bias[ch];
  for (int r = 0; r < 9; ++r) {
    int dy = r / 3 - 1, dx = r % 3 - 1;
    int yy = y + dy, xx = x + dx;
    if (yy < 0 || yy >= H || xx < 0 || xx >= W) continue;
    const __bf16* p = D + (long)((b * H + yy) * W + xx) * inCS + inOff;
    const float* wr = w + (long)ch * Cin * 9 + r;
    for (int c = 0; c < Cin; ++c) acc += (float)p[c] * wr[(long)c * 9];
  }
  out[i] = acc + (up ? up[i] : 0.0f);
}

// ============================================================================
// Host
// ============================================================================
static inline int nblk(long n) { return (int)((n + 255) / 256); }

extern "C" void kernel_launch(void* const* d_in, const int* in_sizes, int n_in,
                              void* d_out, int out_size, void* d_ws, size_t ws_size,
                              hipStream_t stream) {
  // ---- identify inputs by element counts ----
  const float* feat[8] = {0};                      // c13,c14,c15,c16,c23,c24,c25,c26
  const float* Wc[4][5]; const float* Bc[4][5];    // level 0..3 = l6,l5,l4,l3
  const float* Wp[4] = {0}; const float* Bp[4] = {0};
  int levOrder[4] = {0, 1, 2, 3};
  int nlev = 0;
  for (int i = 0; i < n_in; ++i) {
    int s = in_sizes[i];
    int lv = (s == 93312) ? 0 : (s == 1275264) ? 1 : (s == 685440) ? 2 : (s == 390528) ? 3 : -1;
    if (lv >= 0 && nlev < 4) levOrder[nlev++] = lv;
  }
  int c9 = 0, c47 = 0, c23c = 0, c147 = 0, c221 = 0, c129 = 0, c46 = 0, c17 = 0;
  int c128 = 0, c96 = 0, c64 = 0, c32 = 0, c2 = 0;
  for (int i = 0; i < n_in; ++i) {
    const float* p = (const float*)d_in[i];
    switch (in_sizes[i]) {
      case 9437184: feat[c9 == 0 ? 0 : 4] = p; ++c9; break;
      case 4718592: { static const int m[4] = {1, 3, 5, 7}; feat[m[c47 & 3]] = p; ++c47; } break;
      case 2359296: feat[c23c == 0 ? 2 : 6] = p; ++c23c; break;
      case 93312:   Wc[0][0] = p; break;
      case 1275264: Wc[1][0] = p; break;
      case 685440:  Wc[2][0] = p; break;
      case 390528:  Wc[3][0] = p; break;
      case 147456:  Wc[levOrder[c147++ & 3]][1] = p; break;
      case 221184:  Wc[levOrder[c221++ & 3]][2] = p; break;
      case 129024:  Wc[levOrder[c129++ & 3]][3] = p; break;
      case 46080:   Wc[levOrder[c46++  & 3]][4] = p; break;
      case 1728:    Wp[levOrder[c17++  & 3]]    = p; break;
      case 128: { int j = c128++; Bc[levOrder[(j >> 1) & 3]][j & 1] = p; } break;
      case 96:      Bc[levOrder[c96++ & 3]][2] = p; break;
      case 64:      Bc[levOrder[c64++ & 3]][3] = p; break;
      case 32:      Bc[levOrder[c32++ & 3]][4] = p; break;
      case 2:       Bp[levOrder[c2++  & 3]]    = p; break;
      default: break;                               // img_h/img_w (size 1)
    }
  }

  // ---- level configuration (0=l6,1=l5,2=l4,3=l3) ----
  static const int Hs[4]  = {12, 12, 24, 48};
  static const int Wsz[4] = {24, 24, 48, 96};
  static const int Cx[4]  = {2048, 1024, 512, 256};
  static const int ODs[4] = {81, 1107, 595, 339};
  static const int CSs[4] = {96, 1120, 608, 352};   // od padded to mult of 32
  const int Bn = 8;

  // ---- workspace bump allocator ----
  char* ws = (char*)d_ws;
  size_t off = 0;
  auto walloc = [&](size_t bytes) -> void* {
    off = (off + 255) & ~(size_t)255;
    void* r = ws + off;
    off += bytes;
    return r;
  };
  __bf16* XIN  = (__bf16*)walloc((size_t)36864 * 352 * 2);   // largest level-3 staging
  __bf16* DBUF = (__bf16*)walloc((size_t)36864 * 448 * 2);   // decoder feature buffer
  float*  WARP = (float*) walloc((size_t)8 * 256 * 48 * 96 * 4);
  float*  flow6 = (float*)walloc((size_t)8 * 2 * 12 * 24 * 4);
  float*  flow5 = (float*)walloc((size_t)8 * 2 * 12 * 24 * 4);
  float*  up5   = (float*)walloc((size_t)8 * 2 * 24 * 48 * 4);
  float*  flow4 = (float*)walloc((size_t)8 * 2 * 24 * 48 * 4);
  float*  up4   = (float*)walloc((size_t)8 * 2 * 48 * 96 * 4);
  float*  flow3 = (float*)walloc((size_t)8 * 2 * 48 * 96 * 4);
  __bf16* WPK[4][5];
  static const int convCinP[5] = {0, 128, 256, 224, 160};    // [0] per-level
  static const int convCout[5] = {128, 128, 96, 64, 32};
  for (int lv = 0; lv < 4; ++lv)
    for (int j = 0; j < 5; ++j) {
      int cinp = (j == 0) ? CSs[lv] : convCinP[j];
      WPK[lv][j] = (__bf16*)walloc((size_t)9 * cinp * convCout[j] * 2);
    }

  // ---- pack weights (every launch; deterministic) ----
  for (int lv = 0; lv < 4; ++lv)
    for (int j = 0; j < 5; ++j) {
      int cin  = (j == 0) ? ODs[lv] : convCinP[j];
      int cinp = (j == 0) ? CSs[lv] : convCinP[j];
      long total = (long)9 * cinp * convCout[j];
      pack_weights<<<nblk(total), 256, 0, stream>>>(Wc[lv][j], WPK[lv][j],
                                                    cin, cinp, convCout[j] / 16, total);
    }

  auto conv = [&](const __bf16* in, int inCS, int inOff, int lv, int j,
                  __bf16* outp, int outCS, int outOff, int H, int W, int cinp) {
    int pixels = Bn * H * W;
    int mtiles = (pixels + 15) / 16;
    dim3 g((mtiles + 3) / 4), blk(128);
    switch (convCout[j] / 16) {
      case 8: conv3x3_wmma<8><<<g, blk, 0, stream>>>(in, inCS, inOff, WPK[lv][j], Bc[lv][j],
                  outp, outCS, outOff, Bn, H, W, cinp, 1); break;
      case 6: conv3x3_wmma<6><<<g, blk, 0, stream>>>(in, inCS, inOff, WPK[lv][j], Bc[lv][j],
                  outp, outCS, outOff, Bn, H, W, cinp, 1); break;
      case 4: conv3x3_wmma<4><<<g, blk, 0, stream>>>(in, inCS, inOff, WPK[lv][j], Bc[lv][j],
                  outp, outCS, outOff, Bn, H, W, cinp, 1); break;
      default: conv3x3_wmma<2><<<g, blk, 0, stream>>>(in, inCS, inOff, WPK[lv][j], Bc[lv][j],
                  outp, outCS, outOff, Bn, H, W, cinp, 1); break;
    }
  };

  auto do_level = [&](int lv, const float* f1, const float* f2src,
                      const float* upFlow, float* flowOut) {
    const int H = Hs[lv], W = Wsz[lv], C = Cx[lv], CS = CSs[lv];
    const long pixels = (long)Bn * H * W;
    const float* f2 = f2src;
    if (upFlow) {
      long nw = (long)Bn * C * H * W;
      warp_kernel<<<nblk(nw), 256, 0, stream>>>(f2src, upFlow, WARP, Bn, C, H, W);
      f2 = WARP;
    }
    fill_bf16_zero<<<nblk(pixels * CS), 256, 0, stream>>>(XIN, pixels * CS);
    corr_kernel<<<nblk(pixels * 81), 256, 0, stream>>>(f1, f2, XIN, Bn, C, H, W, CS);
    if (upFlow) {
      long nc = (long)Bn * C * H * W;
      nchw_to_nhwc_bf16<<<nblk(nc), 256, 0, stream>>>(f1, XIN, Bn, C, H, W, CS, 81);
      long nf = (long)Bn * 2 * H * W;
      nchw_to_nhwc_bf16<<<nblk(nf), 256, 0, stream>>>(upFlow, XIN, Bn, 2, H, W, CS, 81 + C);
    }
    // decoder: x0..x4 packed at channel offsets {0,128,256,352,416} of DBUF (CS=448)
    conv(XIN,  CS,  0,   lv, 0, DBUF, 448, 0,   H, W, CS);
    conv(DBUF, 448, 0,   lv, 1, DBUF, 448, 128, H, W, 128);
    conv(DBUF, 448, 0,   lv, 2, DBUF, 448, 256, H, W, 256);
    conv(DBUF, 448, 128, lv, 3, DBUF, 448, 352, H, W, 224);
    conv(DBUF, 448, 256, lv, 4, DBUF, 448, 416, H, W, 160);
    long np = (long)Bn * 2 * H * W;
    pred_conv<<<nblk(np), 256, 0, stream>>>(DBUF, Wp[lv], Bp[lv], upFlow, flowOut,
                                            Bn, H, W, 448, 352, 96);
  };

  // ---- pipeline ----
  do_level(0, feat[3], feat[7], nullptr, flow6);              // l6: corr(c16,c26)
  do_level(1, feat[2], feat[6], flow6, flow5);                // l5
  resize_ac_kernel<<<nblk((long)16 * 24 * 48), 256, 0, stream>>>(flow5, up5, 16, 12, 24, 24, 48, 2.0f);
  do_level(2, feat[1], feat[5], up5, flow4);                  // l4
  resize_ac_kernel<<<nblk((long)16 * 48 * 96), 256, 0, stream>>>(flow4, up4, 16, 24, 48, 48, 96, 2.0f);
  do_level(3, feat[0], feat[4], up4, flow3);                  // l3

  // ---- outputs: (flow3*4 -> 384x768, flow4*4 -> 192x384, flow5*4 -> 96x192) ----
  float* out = (float*)d_out;
  resize_ac_kernel<<<nblk((long)16 * 384 * 768), 256, 0, stream>>>(flow3, out, 16, 48, 96, 384, 768, 4.0f);
  resize_ac_kernel<<<nblk((long)16 * 192 * 384), 256, 0, stream>>>(flow4, out + 4718592, 16, 24, 48, 192, 384, 4.0f);
  resize_ac_kernel<<<nblk((long)16 * 96 * 192), 256, 0, stream>>>(flow5, out + 5898240, 16, 12, 24, 96, 192, 4.0f);
}
